// HyperbolicAttention_28492813042259
// MI455X (gfx1250) — matrix operand order repaired and checked by reading, not compiled, not run on hardware
//
#include <hip/hip_runtime.h>

// HyperbolicAttention, fused flash-attention style, CDNA5 (gfx1250, wave32).
// Kernel 1: fp32 -> f16 convert + per-row ||q||^2 and 1/t into d_ws.
// Kernel 2: per-wave 16-row strip, loop over 32-key blocks:
//   - key/value tile staged by the Tensor Data Mover (tensor_load_to_lds,
//     double-buffered, s_wait_tensorcnt) when the builtin is available
//   - S = Q K^T via v_wmma_f32_16x16x32_f16
//   - hyperbolic distance transform in log2 domain (v_log_f32 / v_exp_f32)
//   - online softmax with ds_swizzle xor-reductions
//   - O += P V via WMMA using a transposed LDS image of the key tile.

#define NPTS 8192
#define DDIM 256
#define BM   16
#define BN   32
#define NKB  (NPTS / BN)
#define EPSF 1e-6f

#if defined(__has_builtin)
#if __has_builtin(__builtin_amdgcn_tensor_load_to_lds)
#define USE_TDM 1
#endif
#endif
#ifndef USE_TDM
#define USE_TDM 0
#endif

typedef __attribute__((ext_vector_type(16))) _Float16     v16h;
typedef __attribute__((ext_vector_type(8)))  float        v8f;
typedef __attribute__((ext_vector_type(4)))  unsigned int u32x4;

union Frag {
  v16h  v;
  u32x4 q[2];
};

template <int M>
__device__ __forceinline__ float swz_xor(float x) {
  int i = __builtin_bit_cast(int, x);
  i = __builtin_amdgcn_ds_swizzle(i, (M << 10) | 0x1f);  // group-of-32, xor=M, and=0x1f
  return __builtin_bit_cast(float, i);
}

__device__ __forceinline__ float red_max16(float v) {
  v = fmaxf(v, swz_xor<1>(v));
  v = fmaxf(v, swz_xor<2>(v));
  v = fmaxf(v, swz_xor<4>(v));
  v = fmaxf(v, swz_xor<8>(v));
  return v;
}

__device__ __forceinline__ float red_sum16(float v) {
  v += swz_xor<1>(v);
  v += swz_xor<2>(v);
  v += swz_xor<4>(v);
  v += swz_xor<8>(v);
  return v;
}

#if USE_TDM
// Issue one TDM load of a 32x256 f16 tile (row-major, stride 256) into LDS.
// D# per CDNA5 ISA (2-D tensor, groups 2/3 zero).
__device__ __forceinline__ void tdm_load_tile(unsigned lds_addr, unsigned long long gaddr) {
  typedef __attribute__((ext_vector_type(4))) unsigned int v4u;
  typedef __attribute__((ext_vector_type(8))) int          v8i;
  typedef __attribute__((ext_vector_type(4))) int          v4i;
  v4u g0;
  g0.x = 1u;                                                   // count=1 (valid), user mode
  g0.y = lds_addr;                                             // LDS byte address
  g0.z = (unsigned)(gaddr & 0xffffffffull);                    // global_addr[31:0]
  g0.w = (unsigned)((gaddr >> 32) & 0x1ffffffull) | 0x80000000u; // addr[56:32] | type=2
  v8i g1;
  g1[0] = 0x00010000;          // workgroup_mask=0, data_size=1 (2 bytes)
  g1[1] = (int)(256u << 16);   // tensor_dim0[15:0]=256  (atomic_barrier_addr=0)
  g1[2] = (int)(32u << 16);    // tensor_dim0 hi=0 | tensor_dim1[15:0]=32
  g1[3] = (int)(256u << 16);   // tensor_dim1 hi=0 | tile_dim0=256
  g1[4] = 32;                  // tile_dim1=32 | tile_dim2=0
  g1[5] = 256;                 // tensor_dim0_stride lo = 256
  g1[6] = 0;                   // stride0 hi | tensor_dim1_stride lo
  g1[7] = 0;
  v4i z4 = {};
#if __clang_major__ >= 23
  v8i z8 = {};
  __builtin_amdgcn_tensor_load_to_lds(g0, g1, z4, z4, z8, 0);
#else
  __builtin_amdgcn_tensor_load_to_lds(g0, g1, z4, z4, 0);
#endif
}

__device__ __forceinline__ unsigned lds_off(const void* p) {
  // Flat shared pointer: low 32 bits are the wave-relative LDS byte address.
  return (unsigned)(unsigned long long)(__UINTPTR_TYPE__)p;
}
#endif  // USE_TDM

// ---------------------------------------------------------------------------
// Kernel 1: convert Q to f16, compute qn[i] = ||q_i||^2, invt[i] = 1/max(1-c*qn, eps)
// ---------------------------------------------------------------------------
__global__ void __launch_bounds__(256, 1)
hyp_prep_kernel(const float* __restrict__ q, const float* __restrict__ cptr,
                _Float16* __restrict__ qh, float* __restrict__ qn,
                float* __restrict__ invt)
{
  __shared__ float red[8];
  const int row = blockIdx.x;
  const int tid = threadIdx.x;

  const float v = q[(size_t)row * DDIM + tid];
  qh[(size_t)row * DDIM + tid] = (_Float16)v;

  float s = v * v;
  s += swz_xor<1>(s);
  s += swz_xor<2>(s);
  s += swz_xor<4>(s);
  s += swz_xor<8>(s);
  s += swz_xor<16>(s);
  if ((tid & 31) == 0) red[tid >> 5] = s;
  __syncthreads();
  if (tid == 0) {
    float tot = 0.0f;
#pragma unroll
    for (int i = 0; i < 8; ++i) tot += red[i];
    qn[row] = tot;
    const float cc = fmaxf(cptr[0], EPSF);
    invt[row] = 1.0f / fmaxf(1.0f - cc * tot, EPSF);
  }
}

// ---------------------------------------------------------------------------
// Kernel 2: fused hyperbolic attention (one wave per 16-row query strip)
// ---------------------------------------------------------------------------
__global__ void __launch_bounds__(32, 1)
hyp_attn_kernel(const _Float16* __restrict__ Qh,
                const float* __restrict__ qn,
                const float* __restrict__ invt,
                const float* __restrict__ cptr,
                float* __restrict__ out)
{
  __shared__ __align__(16) _Float16 sQ[BM * DDIM];        //  8 KB query tile
#if USE_TDM
  __shared__ __align__(16) _Float16 sK[2 * BN * DDIM];    // 32 KB key tile, double buffered (TDM)
#else
  __shared__ __align__(16) _Float16 sK[BN * DDIM];        // 16 KB key tile
#endif
  __shared__ __align__(16) _Float16 sKT[DDIM * BN];       // 16 KB key tile, transposed
  __shared__ __align__(16) _Float16 sP[16 * BN];          //  1 KB P bounce (C-layout -> A-layout)

  const int lane = threadIdx.x & 31;
  const int m    = lane & 15;
  const int hi   = lane >> 4;
  const int row0 = blockIdx.x * BM;

  const float cc   = fmaxf(cptr[0], EPSF);
  const float twoC = 2.0f * cc;
  const float isc  = __builtin_amdgcn_rsqf(cc);   // 1/sqrt(c)

  // Stage the query tile (16 rows x 256 f16) into LDS.
  {
    const u32x4* qg = (const u32x4*)(Qh + (size_t)row0 * DDIM);
    u32x4* s4 = (u32x4*)sQ;
#pragma unroll
    for (int i = 0; i < (BM * DDIM / 8) / 32; ++i)
      s4[i * 32 + lane] = qg[i * 32 + lane];
  }

  // Per-row stats (row = row0 + r + 8*hi, replicated across each 16-lane half).
  float qni[8], iti[8];
#pragma unroll
  for (int r = 0; r < 8; ++r) {
    const int rw = row0 + r + 8 * hi;
    qni[r] = qn[rw];
    iti[r] = invt[rw];
  }

  v8f zero8 = {};
  v8f Oacc[16];
#pragma unroll
  for (int n = 0; n < 16; ++n) Oacc[n] = zero8;
  float mrun[8], lrun[8];
#pragma unroll
  for (int r = 0; r < 8; ++r) { mrun[r] = -3.0e38f; lrun[r] = 0.0f; }

#if USE_TDM
  // Prologue: DMA key tile 0 into buffer 0.
  tdm_load_tile(lds_off(sK), (unsigned long long)(__UINTPTR_TYPE__)Qh);
  int buf = 0;
#endif

  asm volatile("s_wait_dscnt 0x0" ::: "memory");
  __syncthreads();

  for (int kb = 0; kb < NKB; ++kb) {
#if USE_TDM
    // Wait for the in-flight TDM tile, then immediately start the next one.
    __builtin_amdgcn_s_wait_tensorcnt(0);
    const _Float16* sKb = sK + buf * (BN * DDIM);
    if (kb + 1 < NKB) {
      tdm_load_tile(lds_off(sK + (buf ^ 1) * (BN * DDIM)),
                    (unsigned long long)(__UINTPTR_TYPE__)(Qh + (size_t)(kb + 1) * BN * DDIM));
    }
    // Build the transposed image from the resident tile (LDS -> LDS).
#pragma unroll
    for (int i = 0; i < (BN * DDIM / 8) / 32; ++i) {
      const int idx = i * 32 + lane;
      u32x4 val = ((const u32x4*)sKb)[idx];
      const int r  = idx >> 5;
      const int c0 = (idx & 31) * 8;
      const _Float16* hv = (const _Float16*)&val;
#pragma unroll
      for (int e = 0; e < 8; ++e) sKT[(c0 + e) * BN + r] = hv[e];
    }
#else
    const _Float16* sKb = sK;
    const _Float16* kg = Qh + (size_t)kb * BN * DDIM;
#pragma unroll
    for (int i = 0; i < (BN * DDIM / 8) / 32; ++i) {
      const int idx = i * 32 + lane;
      u32x4 val = ((const u32x4*)kg)[idx];
      ((u32x4*)sK)[idx] = val;
      const int r  = idx >> 5;
      const int c0 = (idx & 31) * 8;
      const _Float16* hv = (const _Float16*)&val;
#pragma unroll
      for (int e = 0; e < 8; ++e) sKT[(c0 + e) * BN + r] = hv[e];
    }
    {
      const int nb = (kb + 1 < NKB) ? (kb + 1) : kb;
      __builtin_prefetch((const char*)(Qh + (size_t)nb * BN * DDIM) + lane * 512, 0, 1);
    }
#endif
    asm volatile("s_wait_dscnt 0x0" ::: "memory");
    __syncthreads();

    // ---- S = Q K^T : two 16x16 f32 tiles over the 32 keys -------------------
    v8f s0 = zero8, s1 = zero8;
#pragma unroll
    for (int ks = 0; ks < DDIM / 32; ++ks) {
      Frag a, b0, b1;
      // A frag (16x32 f16): lane m holds row m; k = hi*16 + 0..15 contiguous
      const _Float16* qrow = sQ + m * DDIM + ks * 32 + hi * 16;
      a.q[0] = *(const u32x4*)(qrow);
      a.q[1] = *(const u32x4*)(qrow + 8);
      // B frag (32x16 f16): lane m = key column; k = hi*16 + 0..15 contiguous
      const _Float16* kr0 = sKb + m * DDIM + ks * 32 + hi * 16;
      b0.q[0] = *(const u32x4*)(kr0);
      b0.q[1] = *(const u32x4*)(kr0 + 8);
      const _Float16* kr1 = sKb + (m + 16) * DDIM + ks * 32 + hi * 16;
      b1.q[0] = *(const u32x4*)(kr1);
      b1.q[1] = *(const u32x4*)(kr1 + 8);
      s0 = __builtin_amdgcn_wmma_f32_16x16x32_f16(false, a.v, false, b0.v, (short)0, s0, false, false);
      s1 = __builtin_amdgcn_wmma_f32_16x16x32_f16(false, a.v, false, b1.v, (short)0, s1, false, false);
    }

    // ---- hyperbolic transform (log2 domain) --------------------------------
    const int   c0   = kb * BN + m;
    const float qnj0 = qn[c0];
    const float qnj1 = qn[c0 + 16];
    const float ctj0 = twoC * invt[c0];
    const float ctj1 = twoC * invt[c0 + 16];

    float sc0[8], sc1[8];
#pragma unroll
    for (int r = 0; r < 8; ++r) {
      float g  = s0[r];
      float df = fmaxf(qni[r] + qnj0 - 2.0f * g, 0.0f);
      float ar = fmaxf(1.0f + df * iti[r] * ctj0, 1.0f + EPSF) + EPSF;
      float w  = ar + __builtin_amdgcn_sqrtf(ar * ar - 1.0f);
      sc0[r]   = -isc * __builtin_amdgcn_logf(w);   // score / ln2
      g  = s1[r];
      df = fmaxf(qni[r] + qnj1 - 2.0f * g, 0.0f);
      ar = fmaxf(1.0f + df * iti[r] * ctj1, 1.0f + EPSF) + EPSF;
      w  = ar + __builtin_amdgcn_sqrtf(ar * ar - 1.0f);
      sc1[r]   = -isc * __builtin_amdgcn_logf(w);
    }

    // ---- online softmax (per-row state replicated in each 16-lane half) ----
    float alpha[8];
#pragma unroll
    for (int r = 0; r < 8; ++r) {
      const float mx   = red_max16(fmaxf(sc0[r], sc1[r]));
      const float mnew = fmaxf(mrun[r], mx);
      alpha[r] = __builtin_amdgcn_exp2f(mrun[r] - mnew);
      mrun[r]  = mnew;
      const float p0 = __builtin_amdgcn_exp2f(sc0[r] - mnew);
      const float p1 = __builtin_amdgcn_exp2f(sc1[r] - mnew);
      lrun[r] = lrun[r] * alpha[r] + red_sum16(p0 + p1);
      sP[(r + 8 * hi) * BN + m]      = (_Float16)p0;
      sP[(r + 8 * hi) * BN + m + 16] = (_Float16)p1;
    }
    asm volatile("s_wait_dscnt 0x0" ::: "memory");

    // P as A fragment (16 rows x 32 keys).
    Frag pa;
    const _Float16* prow = sP + m * BN + hi * 16;
    pa.q[0] = *(const u32x4*)(prow);
    pa.q[1] = *(const u32x4*)(prow + 8);

    // Rescale running output, then O += P * V.
#pragma unroll
    for (int n = 0; n < 16; ++n) {
#pragma unroll
      for (int r = 0; r < 8; ++r) Oacc[n][r] *= alpha[r];
    }
#pragma unroll
    for (int n = 0; n < 16; ++n) {
      Frag bv;  // B frag of V: lane m = d-column (n*16+m); k = hi*16 + 0..15 from sKT
      const _Float16* vt = sKT + (n * 16 + m) * BN + hi * 16;
      bv.q[0] = *(const u32x4*)(vt);
      bv.q[1] = *(const u32x4*)(vt + 8);
      Oacc[n] = __builtin_amdgcn_wmma_f32_16x16x32_f16(false, pa.v, false, bv.v, (short)0, Oacc[n], false, false);
    }

#if USE_TDM
    buf ^= 1;
#endif
    __syncthreads();
  }

  // ---- normalize and write out ---------------------------------------------
#pragma unroll
  for (int r = 0; r < 8; ++r) lrun[r] = __builtin_amdgcn_rcpf(lrun[r]);
#pragma unroll
  for (int n = 0; n < 16; ++n) {
#pragma unroll
    for (int r = 0; r < 8; ++r) {
      const int rw = row0 + r + 8 * hi;
      out[(size_t)rw * DDIM + n * 16 + m] = Oacc[n][r] * lrun[r];
    }
  }
}

// ---------------------------------------------------------------------------
extern "C" void kernel_launch(void* const* d_in, const int* in_sizes, int n_in,
                              void* d_out, int out_size, void* d_ws, size_t ws_size,
                              hipStream_t stream)
{
  (void)in_sizes; (void)n_in; (void)out_size; (void)ws_size;

  const float* q = (const float*)d_in[0];
  const float* c = (const float*)d_in[1];
  float* out = (float*)d_out;

  char* ws = (char*)d_ws;
  _Float16* qh  = (_Float16*)ws;                                         // 4 MB f16 copy of Q
  float*    qnv = (float*)(ws + (size_t)NPTS * DDIM * sizeof(_Float16)); // 32 KB row norms
  float*    itv = qnv + NPTS;                                            // 32 KB 1/t

  hyp_prep_kernel<<<NPTS, 256, 0, stream>>>(q, c, qh, qnv, itv);
  hyp_attn_kernel<<<NPTS / BM, 32, 0, stream>>>(qh, qnv, itv, c, out);
}
